// GroupedMLP_8959301779837
// MI455X (gfx1250) — matrix operand order repaired and checked by reading, not compiled, not run on hardware
//
#include <hip/hip_runtime.h>
#include <math.h>

// ---------------------------------------------------------------------------
// Grouped MoE MLP for MI455X (gfx1250, wave32).
//   pass 0: x -> bf16; w1,w2 -> bf16 transposed to [E][Ncols][K] (K contiguous)
//   fc1:  x_bf16 @ w1t -> exact-gelu -> inter (bf16)   [v_wmma_f32_16x16x32_bf16]
//   fc2:  inter  @ w2t -> d_out (f32)
// Hot loop: GLOBAL_LOAD_ASYNC_TO_LDS_B128 (saddr form, ASYNCcnt-pipelined)
// into double-buffered LDS; fragments via ds_load_b128 in exact WMMA layouts.
// Block tile 128x128, 8 waves, each wave = 32x64 (2x4 WMMA frags).
// ---------------------------------------------------------------------------

typedef __bf16 v16bf  __attribute__((ext_vector_type(16)));
typedef float  v8f    __attribute__((ext_vector_type(8)));
typedef unsigned int uint4v __attribute__((ext_vector_type(4)));
typedef unsigned int uint2v __attribute__((ext_vector_type(2)));
typedef float float4v __attribute__((ext_vector_type(4)));

struct Frag32B { uint4v lo, hi; };

__device__ __forceinline__ unsigned short f32_to_bf16_rne(float f) {
    unsigned u = __float_as_uint(f);
    u += 0x7FFFu + ((u >> 16) & 1u);
    return (unsigned short)(u >> 16);
}

__device__ __forceinline__ unsigned lds32(const void* p) {
    return (unsigned)(size_t)p;   // flat LDS addr low 32 bits == LDS byte address
}

// Async 16B global -> LDS, GVS addressing: mem = sbase + voff. ASYNCcnt-tracked.
__device__ __forceinline__ void async_b128(unsigned lds_addr, unsigned voff,
                                           unsigned long long sbase) {
    asm volatile("global_load_async_to_lds_b128 %0, %1, %2"
                 :: "v"(lds_addr), "v"(voff), "s"(sbase) : "memory");
}

// A fragment (16x32 bf16): two 16B chunks at k = hi*8 and k = 16+hi*8
__device__ __forceinline__ v16bf load_fragA(const unsigned short* p) {
    Frag32B f;
    f.lo = *(const uint4v*)(p);
    f.hi = *(const uint4v*)(p + 16);
    return __builtin_bit_cast(v16bf, f);
}
// B fragment (32x16 bf16, LDS holds B^T): two 16B chunks at k = hi*8 and k = 16+hi*8
__device__ __forceinline__ v16bf load_fragB(const unsigned short* p) {
    Frag32B f;
    f.lo = *(const uint4v*)(p);
    f.hi = *(const uint4v*)(p + 16);
    return __builtin_bit_cast(v16bf, f);
}

#define TM 128
#define TN 128
#define TK 32
#define LSTR 40   // 80B LDS rows: every 16B chunk stays aligned

// ---------------- pass 0a: elementwise f32 -> bf16 -------------------------
__global__ __launch_bounds__(256)
void cvt_f32_bf16(const float* __restrict__ in, unsigned short* __restrict__ out,
                  long long n4) {
    long long i = (long long)blockIdx.x * blockDim.x + threadIdx.x;
    long long stride = (long long)gridDim.x * blockDim.x;
    for (; i < n4; i += stride) {
        float4v v = ((const float4v*)in)[i];
        uint2v o;
        o.x = (unsigned)f32_to_bf16_rne(v.x) | ((unsigned)f32_to_bf16_rne(v.y) << 16);
        o.y = (unsigned)f32_to_bf16_rne(v.z) | ((unsigned)f32_to_bf16_rne(v.w) << 16);
        ((uint2v*)out)[i] = o;
    }
}

// ---------------- pass 0b: per-expert f32 [K][N] -> bf16 [N][K] ------------
__global__ __launch_bounds__(256)
void transpose_cvt(const float* __restrict__ W, unsigned short* __restrict__ Wt,
                   int K, int Ncols) {
    __shared__ unsigned short tile[32][33];
    const float*    We  = W  + (size_t)blockIdx.z * K * Ncols;
    unsigned short* Wte = Wt + (size_t)blockIdx.z * K * Ncols;
    const int n0 = blockIdx.x * 32, k0 = blockIdx.y * 32;
    const int t = threadIdx.x;
    {   // read 32x32 f32 tile, coalesced along n (float4 per thread)
        int r = t >> 3, c = (t & 7) * 4;
        float4v v = *(const float4v*)&We[(size_t)(k0 + r) * Ncols + (n0 + c)];
        tile[r][c + 0] = f32_to_bf16_rne(v.x);
        tile[r][c + 1] = f32_to_bf16_rne(v.y);
        tile[r][c + 2] = f32_to_bf16_rne(v.z);
        tile[r][c + 3] = f32_to_bf16_rne(v.w);
    }
    __syncthreads();
    {   // write transposed, coalesced along k (4 bf16 = 8B per thread)
        int n = t >> 3, k = (t & 7) * 4;
        uint2v o;
        o.x = (unsigned)tile[k + 0][n] | ((unsigned)tile[k + 1][n] << 16);
        o.y = (unsigned)tile[k + 2][n] | ((unsigned)tile[k + 3][n] << 16);
        *(uint2v*)&Wte[(size_t)(n0 + n) * K + (k0 + k)] = o;
    }
}

// ---------------- grouped GEMM, all-bf16 operands, async-pipelined ---------
template <bool GELU_BF16>
__global__ __launch_bounds__(256, 2)
void grouped_gemm_wmma(const unsigned short* __restrict__ A,   // [N][K] bf16
                       const unsigned short* __restrict__ Wt,  // [E][Ncols][K] bf16
                       void* __restrict__ Outop,
                       const int* __restrict__ tokens,
                       int nExp, int K, int Ncols)
{
    __shared__ __align__(16) unsigned short Alds[2][TM * LSTR];
    __shared__ __align__(16) unsigned short Blds[2][TN * LSTR];

    const int tid   = threadIdx.x;
    const int lane  = tid & 31;
    const int wave  = tid >> 5;
    const int hi    = lane >> 4;
    const int l16   = lane & 15;
    const int waveM = wave >> 1;       // 0..3 -> M offset waveM*32
    const int waveN = wave & 1;        // 0..1 -> N offset waveN*64

    const int rowBase = blockIdx.y * TM;
    const int colBase = blockIdx.x * TN;

    // uniform per-tile expert lookup (tiles don't straddle: 2048 % 128 == 0)
    int e = 0, off = 0;
    for (int i = 0; i < nExp; ++i) {
        int t = tokens[i];
        if (rowBase < off + t) { e = i; break; }
        off += t;
    }

    // ---- async staging: scalar bases advance; per-lane 32-bit offsets fixed
    // Each 128x32 tile = 512 x 16B chunks = 2 per thread (rows r and r+64).
    const int r  = tid >> 2;            // 0..63
    const int ak = (tid & 3) * 8;       // 0,8,16,24 (bf16 elems)
    unsigned long long aBase = (unsigned long long)(size_t)A;
    unsigned long long bBase = (unsigned long long)(size_t)(Wt + (size_t)e * Ncols * K);
    const unsigned aOff0 = (unsigned)(((size_t)(rowBase + r)      * K + ak) * 2);
    const unsigned aOff1 = (unsigned)(((size_t)(rowBase + r + 64) * K + ak) * 2);
    const unsigned bOff0 = (unsigned)(((size_t)(colBase + r)      * K + ak) * 2);
    const unsigned bOff1 = (unsigned)(((size_t)(colBase + r + 64) * K + ak) * 2);

    unsigned aDst0[2], aDst1[2], bDst0[2], bDst1[2];
    #pragma unroll
    for (int b = 0; b < 2; ++b) {
        aDst0[b] = lds32(&Alds[b][r * LSTR + ak]);
        aDst1[b] = lds32(&Alds[b][(r + 64) * LSTR + ak]);
        bDst0[b] = lds32(&Blds[b][r * LSTR + ak]);
        bDst1[b] = lds32(&Blds[b][(r + 64) * LSTR + ak]);
    }

    const unsigned short* aFrag[2];
    const unsigned short* bFrag[2];
    #pragma unroll
    for (int b = 0; b < 2; ++b) {
        aFrag[b] = &Alds[b][(waveM * 32 + l16) * LSTR + hi * 8];
        bFrag[b] = &Blds[b][(waveN * 64 + l16) * LSTR + hi * 8];
    }

    v8f acc[2][4];
    #pragma unroll
    for (int mi = 0; mi < 2; ++mi)
        #pragma unroll
        for (int ni = 0; ni < 4; ++ni) acc[mi][ni] = (v8f){};

    auto issue = [&](int buf, unsigned long long aB, unsigned long long bB) {
        async_b128(aDst0[buf], aOff0, aB);
        async_b128(aDst1[buf], aOff1, aB);
        async_b128(bDst0[buf], bOff0, bB);
        async_b128(bDst1[buf], bOff1, bB);
    };

    auto compute = [&](int cur) {
        const unsigned short* aF = aFrag[cur];
        const unsigned short* bF = bFrag[cur];
        v16bf a0 = load_fragA(aF);
        v16bf a1 = load_fragA(aF + 16 * LSTR);
        v16bf b0 = load_fragB(bF);
        v16bf b1 = load_fragB(bF + 16 * LSTR);
        v16bf b2 = load_fragB(bF + 32 * LSTR);
        v16bf b3 = load_fragB(bF + 48 * LSTR);
        acc[0][0] = __builtin_amdgcn_wmma_f32_16x16x32_bf16(false, a0, false, b0, (short)0, acc[0][0], false, false);
        acc[0][1] = __builtin_amdgcn_wmma_f32_16x16x32_bf16(false, a0, false, b1, (short)0, acc[0][1], false, false);
        acc[0][2] = __builtin_amdgcn_wmma_f32_16x16x32_bf16(false, a0, false, b2, (short)0, acc[0][2], false, false);
        acc[0][3] = __builtin_amdgcn_wmma_f32_16x16x32_bf16(false, a0, false, b3, (short)0, acc[0][3], false, false);
        acc[1][0] = __builtin_amdgcn_wmma_f32_16x16x32_bf16(false, a1, false, b0, (short)0, acc[1][0], false, false);
        acc[1][1] = __builtin_amdgcn_wmma_f32_16x16x32_bf16(false, a1, false, b1, (short)0, acc[1][1], false, false);
        acc[1][2] = __builtin_amdgcn_wmma_f32_16x16x32_bf16(false, a1, false, b2, (short)0, acc[1][2], false, false);
        acc[1][3] = __builtin_amdgcn_wmma_f32_16x16x32_bf16(false, a1, false, b3, (short)0, acc[1][3], false, false);
    };

    const int steps = K / TK;

    // prologue: fill buffer 0 (k = 0)
    issue(0, aBase, bBase);

    // pipelined steady state: issue k=(s+1) into the other buffer, wait for
    // this step's 4 chunks (in-order completion), compute.
    for (int s = 0; s + 1 < steps; ++s) {
        aBase += TK * 2;  // +64 bytes
        bBase += TK * 2;
        const int cur = s & 1;
        issue(cur ^ 1, aBase, bBase);
        asm volatile("s_wait_asynccnt 0x4" ::: "memory");
        __syncthreads();
        compute(cur);
        __syncthreads();
    }
    // tail
    asm volatile("s_wait_asynccnt 0x0" ::: "memory");
    __syncthreads();
    compute((steps - 1) & 1);

    // ---- epilogue: C/D layout VGPR r -> M = hi*8 + r, N = l16 ----
    unsigned short* outBf  = (unsigned short*)Outop;
    float*          outF32 = (float*)Outop;
    #pragma unroll
    for (int mi = 0; mi < 2; ++mi) {
        #pragma unroll
        for (int ni = 0; ni < 4; ++ni) {
            const v8f& c = acc[mi][ni];
            int col = colBase + waveN * 64 + ni * 16 + l16;
            #pragma unroll
            for (int rr = 0; rr < 8; ++rr) {
                int row = rowBase + waveM * 32 + mi * 16 + hi * 8 + rr;
                float v = c[rr];
                if constexpr (GELU_BF16) {
                    v = 0.5f * v * (1.0f + erff(v * 0.70710678118654752f));
                    outBf[(size_t)row * Ncols + col] = f32_to_bf16_rne(v);
                } else {
                    outF32[(size_t)row * Ncols + col] = v;
                }
            }
        }
    }
}

extern "C" void kernel_launch(void* const* d_in, const int* in_sizes, int n_in,
                              void* d_out, int out_size, void* d_ws, size_t ws_size,
                              hipStream_t stream) {
    const float* x      = (const float*)d_in[0];   // [N, H] f32
    const float* w1     = (const float*)d_in[1];   // [E, H, F] f32
    const float* w2     = (const float*)d_in[2];   // [E, F, H] f32
    const int*   tokens = (const int*)d_in[3];     // [E] i32

    const int E = in_sizes[3];
    const int H = 1024;
    const int F = 4096;
    const int N = in_sizes[0] / H;                 // 16384

    // workspace layout (bf16 elements): xb | w1t | w2t | inter  (~288 MB)
    unsigned short* wsu   = (unsigned short*)d_ws;
    unsigned short* xb    = wsu;
    unsigned short* w1t   = xb  + (size_t)N * H;
    unsigned short* w2t   = w1t + (size_t)E * H * F;
    unsigned short* inter = w2t + (size_t)E * F * H;

    dim3 blk(256);

    // pass 0: convert + transpose
    cvt_f32_bf16<<<8192, blk, 0, stream>>>(x, xb, (long long)N * H / 4);
    transpose_cvt<<<dim3(F / 32, H / 32, E), blk, 0, stream>>>(w1, w1t, H, F);
    transpose_cvt<<<dim3(H / 32, F / 32, E), blk, 0, stream>>>(w2, w2t, F, H);

    // fc1 + gelu -> inter (bf16)
    grouped_gemm_wmma<true><<<dim3(F / TN, N / TM), blk, 0, stream>>>(
        xb, w1t, inter, tokens, E, H, F);
    // fc2 -> d_out (f32)
    grouped_gemm_wmma<false><<<dim3(H / TN, N / TM), blk, 0, stream>>>(
        inter, w2t, d_out, tokens, E, F, H);
}
